// NTC_64424509440753
// MI455X (gfx1250) — compile-verified, matrix operand order checked by reading, not run on hardware
//
#include <hip/hip_runtime.h>

// Reference: BATCH=4096, RANK=32, 5 stride-2 2x2x2 conv layers on a rank-1
// 32^3 cube, then a 32->1 sigmoid head.
//
// Strategy (MI455X): 1 workgroup (256 thr = 8 wave32) per batch element.
// All intermediates live in LDS (<64KB static). Convs 1..3 are im2col GEMMs
// (M=32 ch, K=256=32ic*8taps, N=spatial) on v_wmma_f32_16x16x32_f16.
// Layer-0 activation is computed on the fly from the separable u (x) i (x) t
// cube and fused into layer 1 as K-halves (slab = 16ch x 2x16x16 f16 = 16KB).
// Layer-0 is strength-reduced per row: out(w0) = bias + cA*t[2w0] + cB*t[2w0+1]
// (2 FMA/value instead of 8), with rows written as 2x ds_store_b128.
// HBM traffic: ~1.7MB total (inputs+weights+out) vs ~2.6GB if intermediates
// spilled to HBM -- the whole game is keeping them in the 320KB WGP LDS.
//
// Input order (setup_inputs dict, lists flattened):
//  0:user(4096x32) 1:item 2:time 3:w0(32x1x8) 4..7:w1..w4(32x32x8)
//  8..12:b0..b4(32) 13:score_w(32) 14:score_b(1)

#define NEG_SLOPE 0.01f

typedef __attribute__((ext_vector_type(16))) _Float16 v16h;
typedef __attribute__((ext_vector_type(8)))  _Float16 v8h;
typedef __attribute__((ext_vector_type(8)))  float    v8f;

// v16h element j -> K offset within a 32-wide K chunk, per the CDNA5 16-bit
// fragment layout (lanes 0-15: K {0..7,16..23}; lanes 16-31: K {8..15,24..31}).
__device__ __forceinline__ int koff(int j, int hi) {
  return j + ((j >= 8) ? 8 : 0) + (hi ? 8 : 0);
}

__device__ __forceinline__ float leaky(float x) {
  return x >= 0.0f ? x : NEG_SLOPE * x;
}

__global__ void __launch_bounds__(256)
ntc_fused(const float* __restrict__ U,  const float* __restrict__ I,
          const float* __restrict__ T,
          const float* __restrict__ W0, const float* __restrict__ W1,
          const float* __restrict__ W2, const float* __restrict__ W3,
          const float* __restrict__ W4,
          const float* __restrict__ B0, const float* __restrict__ B1,
          const float* __restrict__ B2, const float* __restrict__ B3,
          const float* __restrict__ B4,
          const float* __restrict__ SW, const float* __restrict__ SB,
          float* __restrict__ OUT)
{
  __shared__ float   s_u[32], s_i[32], s_t[32];
  __shared__ float   s_w0[256], s_b0[32];
  __shared__ _Float16 s_act0[16 * 512];   // 16ch x 2 x 16 x 16 slab   (16KB)
  __shared__ _Float16 s_act1[32 * 512];   // 32ch x 8x8x8              (32KB)
  __shared__ _Float16 s_act2[32 * 64];    // 32ch x 4x4x4              (4KB)
  __shared__ _Float16 s_act3[32 * 8];     // 32ch x 2x2x2              (0.5KB)

  const int n    = blockIdx.x;
  const int tid  = threadIdx.x;
  const int lane = tid & 31;              // wave32
  const int wid  = tid >> 5;              // 8 waves
  const int hi   = (lane >> 4) & 1;
  const int ln   = lane & 15;
  const int mt   = wid >> 2;              // M tile: channels [16*mt, 16*mt+16)
  const int nt   = wid & 3;               // N tile

  if (tid < 32)       s_u[tid]      = U[n * 32 + tid];
  else if (tid < 64)  s_i[tid - 32] = I[n * 32 + (tid - 32)];
  else if (tid < 96)  s_t[tid - 64] = T[n * 32 + (tid - 64)];
  s_w0[tid] = W0[tid & 255];
  if (tid >= 96 && tid < 128) s_b0[tid - 96] = B0[tid - 96];
  __syncthreads();

  // ---- hoist layer-1 A fragments (f32 weights -> f16 regs, reused 8x) ----
  // K ordering within an ic-half: Kh = icl*8 + k, icl in [0,16), k in [0,8)
  v16h a1[2][4];
  {
    const int c = mt * 16 + ln;
    #pragma unroll
    for (int ich = 0; ich < 2; ++ich)
      #pragma unroll
      for (int kc = 0; kc < 4; ++kc) {
        v16h a = {};
        #pragma unroll
        for (int j = 0; j < 16; ++j) {
          const int Kh = kc * 32 + koff(j, hi);
          const int ic = ich * 16 + (Kh >> 3);
          a[j] = (_Float16)W1[c * 256 + ic * 8 + (Kh & 7)];
        }
        a1[ich][kc] = a;
      }
  }

  // ================= layer 0 (on-the-fly) fused with layer 1 =================
  for (int d1 = 0; d1 < 8; ++d1) {
    v8f acc = {};
    #pragma unroll
    for (int ich = 0; ich < 2; ++ich) {
      __syncthreads();  // previous slab fully consumed before overwrite
      // Build act0 slab: channels [16*ich, 16*ich+16), D0 in {2d1, 2d1+1}.
      // Layout [icl][dd][h0][w0]; one 16-value w0-row per thread iteration.
      // Separable: val(w0) = bias + cA*t[2w0] + cB*t[2w0+1], cA/cB folded once.
      #pragma unroll
      for (int r = 0; r < 2; ++r) {
        const int rid = tid + 256 * r;        // rid = icl*32 + dd*16 + h0
        const int icl = rid >> 5;
        const int dd  = (rid >> 4) & 1;
        const int h0  = rid & 15;
        const int ic  = ich * 16 + icl;
        const int D0  = 2 * d1 + dd;
        const float u0 = s_u[2 * D0], u1 = s_u[2 * D0 + 1];
        const float i0 = s_i[2 * h0], i1 = s_i[2 * h0 + 1];
        const float* wc = s_w0 + ic * 8;
        const float cA = u0 * (i0 * wc[0] + i1 * wc[2]) + u1 * (i0 * wc[4] + i1 * wc[6]);
        const float cB = u0 * (i0 * wc[1] + i1 * wc[3]) + u1 * (i0 * wc[5] + i1 * wc[7]);
        const float bias = s_b0[ic];
        v8h lo, hiq;
        #pragma unroll
        for (int w0v = 0; w0v < 8; ++w0v) {
          lo[w0v]  = (_Float16)leaky(bias + cA * s_t[2 * w0v]      + cB * s_t[2 * w0v + 1]);
          hiq[w0v] = (_Float16)leaky(bias + cA * s_t[2 * w0v + 16] + cB * s_t[2 * w0v + 17]);
        }
        *(v8h*)(s_act0 + rid * 16)     = lo;   // 32B-aligned -> ds_store_b128
        *(v8h*)(s_act0 + rid * 16 + 8) = hiq;
      }
      __syncthreads();

      // layer-1 GEMM for this D1 slice / ic-half: M=32, N=64 (8x8), K=128
      const int np = nt * 16 + ln;        // (h1,w1) position
      const int h1 = np >> 3, w1 = np & 7;
      #pragma unroll
      for (int kc = 0; kc < 4; ++kc) {
        v16h b = {};
        #pragma unroll
        for (int j = 0; j < 16; ++j) {
          const int Kh  = kc * 32 + koff(j, hi);
          const int icl = Kh >> 3;
          const int k   = Kh & 7;
          const int aa = (k >> 2) & 1, bb = (k >> 1) & 1, gg = k & 1;
          b[j] = s_act0[icl * 512 + aa * 256 + (2 * h1 + bb) * 16 + (2 * w1 + gg)];
        }
        acc = __builtin_amdgcn_wmma_f32_16x16x32_f16(
                false, a1[ich][kc], false, b, (short)0, acc, false, false);
      }
    }
    // epilogue -> act1[c][d1*64 + h1*8 + w1]
    const int np = nt * 16 + ln;
    #pragma unroll
    for (int v = 0; v < 8; ++v) {
      const int c = mt * 16 + v + hi * 8;
      s_act1[c * 512 + d1 * 64 + np] = (_Float16)leaky(acc[v] + B1[c]);
    }
  }
  __syncthreads();

  // ================= layer 2: M=32, N=64 (4x4x4), K=256 =================
  {
    v8f acc = {};
    const int np = nt * 16 + ln;
    const int d2 = np >> 4, h2 = (np >> 2) & 3, w2 = np & 3;
    const int c  = mt * 16 + ln;
    for (int kc = 0; kc < 8; ++kc) {
      v16h a = {}, b = {};
      #pragma unroll
      for (int j = 0; j < 16; ++j) {
        const int K = kc * 32 + koff(j, hi);          // K = ic*8 + k
        a[j] = (_Float16)W2[c * 256 + K];
        const int ic = K >> 3, k = K & 7;
        const int aa = (k >> 2) & 1, bb = (k >> 1) & 1, gg = k & 1;
        b[j] = s_act1[ic * 512 + (2 * d2 + aa) * 64 + (2 * h2 + bb) * 8 + (2 * w2 + gg)];
      }
      acc = __builtin_amdgcn_wmma_f32_16x16x32_f16(
              false, a, false, b, (short)0, acc, false, false);
    }
    __syncthreads();
    #pragma unroll
    for (int v = 0; v < 8; ++v) {
      const int cc = mt * 16 + v + hi * 8;
      s_act2[cc * 64 + np] = (_Float16)leaky(acc[v] + B2[cc]);
    }
  }
  __syncthreads();

  // ============ layer 3: M=32, N=8 valid (2x2x2), K=256; waves 0..1 ============
  if (wid < 2) {
    v8f acc = {};
    const int np = ln & 7;                 // columns 8..15 duplicate (discarded)
    const int d3 = np >> 2, h3 = (np >> 1) & 1, w3 = np & 1;
    const int c  = wid * 16 + ln;
    for (int kc = 0; kc < 8; ++kc) {
      v16h a = {}, b = {};
      #pragma unroll
      for (int j = 0; j < 16; ++j) {
        const int K = kc * 32 + koff(j, hi);
        a[j] = (_Float16)W3[c * 256 + K];
        const int ic = K >> 3, k = K & 7;
        const int aa = (k >> 2) & 1, bb = (k >> 1) & 1, gg = k & 1;
        b[j] = s_act2[ic * 64 + (2 * d3 + aa) * 16 + (2 * h3 + bb) * 4 + (2 * w3 + gg)];
      }
      acc = __builtin_amdgcn_wmma_f32_16x16x32_f16(
              false, a, false, b, (short)0, acc, false, false);
    }
    if (ln < 8) {
      #pragma unroll
      for (int v = 0; v < 8; ++v) {
        const int cc = wid * 16 + v + hi * 8;
        s_act3[cc * 8 + np] = (_Float16)leaky(acc[v] + B3[cc]);
      }
    }
  }
  __syncthreads();

  // ============ layer 4 (32 outputs, K=256) + score head: wave 0 ============
  if (wid == 0) {
    const int c = lane;                    // output channel
    float acc4 = B4[c];
    #pragma unroll 8
    for (int q = 0; q < 256; ++q)          // q = ic*8 + k; act3 flat index == q
      acc4 += W4[c * 256 + q] * (float)s_act3[q];
    acc4 = leaky(acc4);
    float red = acc4 * SW[c];
    #pragma unroll
    for (int off = 16; off > 0; off >>= 1)
      red += __shfl_down(red, off, 32);
    if (lane == 0)
      OUT[n] = 1.0f / (1.0f + __expf(-(red + SB[0])));
  }
}

extern "C" void kernel_launch(void* const* d_in, const int* in_sizes, int n_in,
                              void* d_out, int out_size, void* d_ws, size_t ws_size,
                              hipStream_t stream) {
  (void)in_sizes; (void)n_in; (void)d_ws; (void)ws_size;
  const float* U  = (const float*)d_in[0];
  const float* I  = (const float*)d_in[1];
  const float* T  = (const float*)d_in[2];
  const float* W0 = (const float*)d_in[3];
  const float* W1 = (const float*)d_in[4];
  const float* W2 = (const float*)d_in[5];
  const float* W3 = (const float*)d_in[6];
  const float* W4 = (const float*)d_in[7];
  const float* B0 = (const float*)d_in[8];
  const float* B1 = (const float*)d_in[9];
  const float* B2 = (const float*)d_in[10];
  const float* B3 = (const float*)d_in[11];
  const float* B4 = (const float*)d_in[12];
  const float* SW = (const float*)d_in[13];
  const float* SB = (const float*)d_in[14];
  // out_size == BATCH == 4096 workgroups, 256 threads (8 wave32) each.
  ntc_fused<<<out_size, 256, 0, stream>>>(U, I, T, W0, W1, W2, W3, W4,
                                          B0, B1, B2, B3, B4, SW, SB,
                                          (float*)d_out);
}
